// AttentionBlock_89979564851447
// MI455X (gfx1250) — matrix operand order repaired and checked by reading, not compiled, not run on hardware
//
#include <hip/hip_runtime.h>
#include <hip/hip_fp16.h>

// ---------------------------------------------------------------------------
// AttentionBlock for MI455X (gfx1250), wave32 + v_wmma_f32_16x16x32_f16.
//
// Shapes: B=4, H=W=64 (N=4096 pixels/batch), C=256.
// Pipeline:
//   prep : x -> f16, weights -> transposed f16 (Wt[n][k])
//   qkv  : Q = (x@wq+bq)*C^-0.5 (f16), K = x@wk+bk (f16), V stored transposed
//          Vt[b][c][n] (f16) so P*V WMMA A-operand loads are contiguous.
//   flash: per wave: 16 queries; keys staged in LDS in 64-key chunks by the
//          4-wave workgroup via gfx1250 async global->LDS loads (ASYNCcnt).
//          S^T = K*Q^T so online softmax is per-lane, P repacked in-register
//          into the next WMMA's B layout, O^T += V^T * P^T.
//   proj : out = O@wp + bp + x (f32).
// ---------------------------------------------------------------------------

typedef __attribute__((ext_vector_type(16))) _Float16 v16h;
typedef __attribute__((ext_vector_type(8)))  _Float16 v8h;
typedef __attribute__((ext_vector_type(8)))  float    v8f;
typedef __attribute__((ext_vector_type(4)))  unsigned int v4u;
typedef __attribute__((ext_vector_type(4)))  int v4i;

#define AS1 __attribute__((address_space(1)))
#define AS3 __attribute__((address_space(3)))

#if defined(__AMDGCN__) && __has_builtin(__builtin_amdgcn_global_load_async_to_lds_b128)
#define HAVE_ASYNC_LDS 1
#else
#define HAVE_ASYNC_LDS 0
#endif

__device__ __forceinline__ void wait_asynccnt0() {
#if __has_builtin(__builtin_amdgcn_s_wait_asynccnt)
  __builtin_amdgcn_s_wait_asynccnt(0);
#else
  asm volatile("s_wait_asynccnt 0" ::: "memory");
#endif
}

#if HAVE_ASYNC_LDS
// One 16-byte async global->LDS copy (gfx1250 GLOBAL_LOAD_ASYNC_TO_LDS_B128,
// tracked by ASYNCcnt; data bypasses the VGPR file).
__device__ __forceinline__ void async_copy_b128(const _Float16* g, _Float16* l) {
  __builtin_amdgcn_global_load_async_to_lds_b128(
      (AS1 v4i*)(g), (AS3 v4i*)(l), 0, 0);
}
#endif

__device__ __forceinline__ v8f wmma_f16(v16h a, v16h b, v8f c) {
  // (neg_a, A, neg_b, B, c_mod, C, reuse_a, reuse_b)
  return __builtin_amdgcn_wmma_f32_16x16x32_f16(false, a, false, b, (short)0, c,
                                                false, false);
}

// A operand, 16(M) x 32(K), source row-major [m][k], stride ld halves.
// lanes 0-15 row M=lane: halves 0..15 = K0..15;
// lanes 16-31 same M, K16..31.
__device__ __forceinline__ v16h load_fragA(const _Float16* p, int ld) {
  const int lane = threadIdx.x & 31;
  const int r    = lane & 15;
  const int k0   = (lane >> 4) << 4;  // 0 or 16
  const _Float16* row = p + r * ld + k0;
  v8h lo = *(const v8h*)(row);        // K = k0 .. k0+7
  v8h hi = *(const v8h*)(row + 8);    // K = k0+8 .. k0+15
  v16h f;
#pragma unroll
  for (int i = 0; i < 8; ++i) { f[i] = lo[i]; f[i + 8] = hi[i]; }
  return f;
}

// B operand, 32(K) x 16(N), source row-major [n][k] (i.e. transposed weight /
// Q rows), stride ld halves. Lane n<16: halves = K0..15; lane n+16: K16..31.
__device__ __forceinline__ v16h load_fragB(const _Float16* p, int ld) {
  const int lane = threadIdx.x & 31;
  const int r    = lane & 15;
  const int k0   = (lane >> 4) << 4;  // 0 or 16
  const _Float16* row = p + r * ld + k0;
  v8h lo = *(const v8h*)(row);
  v8h hi = *(const v8h*)(row + 8);
  v16h f;
#pragma unroll
  for (int i = 0; i < 8; ++i) { f[i] = lo[i]; f[i + 8] = hi[i]; }
  return f;
}

// ---------------------------------------------------------------------------
__global__ void prep_kernel(const float* __restrict__ x,
                            const float* __restrict__ wq, const float* __restrict__ wk,
                            const float* __restrict__ wv, const float* __restrict__ wp,
                            _Float16* __restrict__ xh, _Float16* __restrict__ WT) {
  int idx = blockIdx.x * blockDim.x + threadIdx.x;
  const int NX = 16384 * 256;
  if (idx < NX) { xh[idx] = (_Float16)x[idx]; return; }
  idx -= NX;
  if (idx >= 4 * 65536) return;
  const int w = idx >> 16, r = idx & 65535;
  const int k = r >> 8, n = r & 255;
  const float* src = (w == 0) ? wq : (w == 1) ? wk : (w == 2) ? wv : wp;
  WT[w * 65536 + n * 256 + k] = (_Float16)src[k * 256 + n];
}

// ---------------------------------------------------------------------------
// Fused QKV projection: one wave computes a 16x64 output tile for q,k,v.
// grid 1024 x 128 threads (4 waves) -> 4096 waves = 1024 row-tiles x 4 col-grp
__global__ __launch_bounds__(128)
void qkv_kernel(const _Float16* __restrict__ xh,
                const _Float16* __restrict__ WqT, const _Float16* __restrict__ WkT,
                const _Float16* __restrict__ WvT,
                const float* __restrict__ bq, const float* __restrict__ bk,
                const float* __restrict__ bv,
                _Float16* __restrict__ Qh, _Float16* __restrict__ Kh,
                _Float16* __restrict__ Vt) {
  const int wave = threadIdx.x >> 5;
  const int gid  = blockIdx.x * 4 + wave;
  const int mt   = gid >> 2;        // 0..1023 row tile
  const int nt   = gid & 3;         // 64-wide column group
  const int m0   = mt * 16;
  const int lane = threadIdx.x & 31;

  v8f dq[4], dk[4], dv[4];
#pragma unroll
  for (int s = 0; s < 4; ++s) { dq[s] = (v8f)0.0f; dk[s] = (v8f)0.0f; dv[s] = (v8f)0.0f; }

  for (int cc = 0; cc < 8; ++cc) {
    v16h a = load_fragA(xh + (size_t)m0 * 256 + cc * 32, 256);
#pragma unroll
    for (int s = 0; s < 4; ++s) {
      const int n0 = nt * 64 + s * 16;
      dq[s] = wmma_f16(a, load_fragB(WqT + n0 * 256 + cc * 32, 256), dq[s]);
      dk[s] = wmma_f16(a, load_fragB(WkT + n0 * 256 + cc * 32, 256), dk[s]);
      dv[s] = wmma_f16(a, load_fragB(WvT + n0 * 256 + cc * 32, 256), dv[s]);
    }
  }

  const int rbase = m0 + ((lane >> 4) << 3);  // C/D: lanes>=16 hold M=j+8
  const int ncol  = lane & 15;
#pragma unroll
  for (int s = 0; s < 4; ++s) {
    const int col = nt * 64 + s * 16 + ncol;
    const float bqv = bq[col], bkv = bk[col], bvv = bv[col];
#pragma unroll
    for (int j = 0; j < 8; ++j) {
      const int r = rbase + j;
      Qh[(size_t)r * 256 + col] = (_Float16)((dq[s][j] + bqv) * 0.0625f); // *C^-0.5
      Kh[(size_t)r * 256 + col] = (_Float16)(dk[s][j] + bkv);
      const int b = r >> 12, n = r & 4095;
      Vt[((size_t)(b * 256 + col)) * 4096 + n] = (_Float16)(dv[s][j] + bvv);
    }
  }
}

// ---------------------------------------------------------------------------
// Flash attention. grid (64, B) x 128 threads. Wave w: queries
// q0 = blockIdx.x*64 + w*16 of batch blockIdx.y. K/V staged in LDS per
// workgroup in 64-key chunks (~70 KB of the 320 KB WGP LDS), shared by all
// 4 waves, via gfx1250 async global->LDS b128 loads when available.
#define QK_LD 264   // 256 + 8 halves pad  (528 B rows: 16B aligned, bank-conflict free)
#define VT_LD 72    // 64 + 8 halves pad   (144 B rows)

__global__ __launch_bounds__(128)
void flash_kernel(const _Float16* __restrict__ Qh, const _Float16* __restrict__ Kh,
                  const _Float16* __restrict__ Vt, _Float16* __restrict__ Oh) {
  __shared__ _Float16 Kl[64 * QK_LD];    // 64 keys x 256 ch
  __shared__ _Float16 Vl[256 * VT_LD];   // 256 ch  x 64 keys (V^T)

  const int b    = blockIdx.y;
  const int wave = threadIdx.x >> 5;
  const int lane = threadIdx.x & 31;
  const int q0   = blockIdx.x * 64 + wave * 16;
  const size_t qrow0 = (size_t)b * 4096 + q0;

  // Q^T B-fragments for all 8 C-chunks (reused for all 4096 keys).
  v16h qf[8];
#pragma unroll
  for (int cc = 0; cc < 8; ++cc)
    qf[cc] = load_fragB(Qh + qrow0 * 256 + cc * 32, 256);

  v8f acc[16];                       // O^T: 16 channel tiles x (16ch x 16q)
#pragma unroll
  for (int t = 0; t < 16; ++t) acc[t] = (v8f)0.0f;
  float mrun = -__builtin_inff(), lrun = 0.0f;

  const int tid = threadIdx.x;
  for (int kk = 0; kk < 4096; kk += 64) {
    __syncthreads();
#if HAVE_ASYNC_LDS
    // stage K rows: 64 x 256 halves (2048 async b128, no VGPR round-trip)
#pragma unroll
    for (int i = 0; i < 16; ++i) {
      const int idx = tid + i * 128;
      const int row = idx >> 5, c8 = (idx & 31) << 3;
      async_copy_b128(Kh + ((size_t)b * 4096 + kk + row) * 256 + c8,
                      &Kl[row * QK_LD + c8]);
    }
    // stage V^T rows: 256 x 64 halves (2048 async b128)
#pragma unroll
    for (int i = 0; i < 16; ++i) {
      const int idx = tid + i * 128;
      const int row = idx >> 3, c8 = (idx & 7) << 3;
      async_copy_b128(Vt + ((size_t)b * 256 + row) * 4096 + kk + c8,
                      &Vl[row * VT_LD + c8]);
    }
    wait_asynccnt0();
#else
#pragma unroll
    for (int i = 0; i < 16; ++i) {
      const int idx = tid + i * 128;
      const int row = idx >> 5, c8 = (idx & 31) << 3;
      *(v8h*)&Kl[row * QK_LD + c8] =
          *(const v8h*)(Kh + ((size_t)b * 4096 + kk + row) * 256 + c8);
    }
#pragma unroll
    for (int i = 0; i < 16; ++i) {
      const int idx = tid + i * 128;
      const int row = idx >> 3, c8 = (idx & 7) << 3;
      *(v8h*)&Vl[row * VT_LD + c8] =
          *(const v8h*)(Vt + ((size_t)b * 256 + row) * 4096 + kk + c8);
    }
#endif
    __syncthreads();

    // S^T = K * Q^T : four 16-key x 16-query tiles, GEMM K-dim = C = 256.
    v8f d[4];
#pragma unroll
    for (int st = 0; st < 4; ++st) d[st] = (v8f)0.0f;
#pragma unroll
    for (int cc = 0; cc < 8; ++cc) {
      const v16h q = qf[cc];
#pragma unroll
      for (int st = 0; st < 4; ++st)
        d[st] = wmma_f16(load_fragA(&Kl[(st * 16) * QK_LD + cc * 32], QK_LD), q, d[st]);
    }

    // Online softmax over this 64-key chunk. Lane holds 32 scores of its
    // query column; partner lane (xor 16) holds the other 32 keys.
    float cmax = d[0][0];
#pragma unroll
    for (int st = 0; st < 4; ++st)
#pragma unroll
      for (int j = 0; j < 8; ++j) cmax = fmaxf(cmax, d[st][j]);
    cmax = fmaxf(cmax, __shfl_xor(cmax, 16));
    const float mnew  = fmaxf(mrun, cmax);
    const float alpha = __expf(mrun - mnew);

    union PU { v8h h; v4u u; } p[4], o[4];
    float psum = 0.0f;
#pragma unroll
    for (int st = 0; st < 4; ++st)
#pragma unroll
      for (int j = 0; j < 8; ++j) {
        const float e = __expf(d[st][j] - mnew);
        psum += e;
        p[st].h[j] = (_Float16)e;
      }
    psum += __shfl_xor(psum, 16);
    lrun = lrun * alpha + psum;
    mrun = mnew;

    // Repack P^T into B-operand layout for the PV WMMAs:
    // pb0 covers keys kk..kk+31 (tiles 0/1), pb1 keys kk+32..kk+63 (tiles 2/3).
    // lane<16 needs the first 16 keys of its half (own rows + partner rows),
    // lane>=16 the second 16 (partner rows + own rows).
#pragma unroll
    for (int st = 0; st < 4; ++st)
#pragma unroll
      for (int i = 0; i < 4; ++i)
        o[st].u[i] = (unsigned)__shfl_xor((int)p[st].u[i], 16);
    const bool lolane = lane < 16;
    v16h pb0, pb1;
#pragma unroll
    for (int j = 0; j < 8; ++j) {
      pb0[j]     = lolane ? p[0].h[j] : o[1].h[j];
      pb0[j + 8] = lolane ? o[0].h[j] : p[1].h[j];
      pb1[j]     = lolane ? p[2].h[j] : o[3].h[j];
      pb1[j + 8] = lolane ? o[2].h[j] : p[3].h[j];
    }

    // O^T = alpha * O^T + V^T * P^T  (16 channel tiles, 2 x K=32 keys)
#pragma unroll
    for (int t = 0; t < 16; ++t) {
#pragma unroll
      for (int j = 0; j < 8; ++j) acc[t][j] *= alpha;
      acc[t] = wmma_f16(load_fragA(&Vl[(t * 16) * VT_LD + 0],  VT_LD), pb0, acc[t]);
      acc[t] = wmma_f16(load_fragA(&Vl[(t * 16) * VT_LD + 32], VT_LD), pb1, acc[t]);
    }
  }

  // Normalize and store O (f16, row-major [pixel][channel]).
  const float inv = 1.0f / lrun;
  const int qcol  = lane & 15;
  const int ch8   = (lane >> 4) << 3;
#pragma unroll
  for (int t = 0; t < 16; ++t) {
    v8h hv;
#pragma unroll
    for (int j = 0; j < 8; ++j) hv[j] = (_Float16)(acc[t][j] * inv);
    *(v8h*)(Oh + (qrow0 + qcol) * 256 + t * 16 + ch8) = hv;
  }
}

// ---------------------------------------------------------------------------
// Output projection + bias + residual: out = O@wp + bp + x   (f32 out)
__global__ __launch_bounds__(128)
void proj_kernel(const _Float16* __restrict__ Oh, const _Float16* __restrict__ WpT,
                 const float* __restrict__ bp, const float* __restrict__ x,
                 float* __restrict__ out) {
  const int wave = threadIdx.x >> 5;
  const int gid  = blockIdx.x * 4 + wave;
  const int mt   = gid >> 2, nt = gid & 3;
  const int m0   = mt * 16;
  const int lane = threadIdx.x & 31;

  v8f d[4];
#pragma unroll
  for (int s = 0; s < 4; ++s) d[s] = (v8f)0.0f;

  for (int cc = 0; cc < 8; ++cc) {
    v16h a = load_fragA(Oh + (size_t)m0 * 256 + cc * 32, 256);
#pragma unroll
    for (int s = 0; s < 4; ++s)
      d[s] = wmma_f16(a, load_fragB(WpT + (nt * 64 + s * 16) * 256 + cc * 32, 256), d[s]);
  }

  const int rbase = m0 + ((lane >> 4) << 3);
  const int ncol  = lane & 15;
#pragma unroll
  for (int s = 0; s < 4; ++s) {
    const int col = nt * 64 + s * 16 + ncol;
    const float bb = bp[col];
#pragma unroll
    for (int j = 0; j < 8; ++j) {
      const int r = rbase + j;
      out[(size_t)r * 256 + col] = d[s][j] + bb + x[(size_t)r * 256 + col];
    }
  }
}

// ---------------------------------------------------------------------------
extern "C" void kernel_launch(void* const* d_in, const int* in_sizes, int n_in,
                              void* d_out, int out_size, void* d_ws, size_t ws_size,
                              hipStream_t stream) {
  const float* x  = (const float*)d_in[0];
  // d_in[1] = t (unused by reference)
  const float* wq = (const float*)d_in[2];
  const float* bq = (const float*)d_in[3];
  const float* wk = (const float*)d_in[4];
  const float* bk = (const float*)d_in[5];
  const float* wv = (const float*)d_in[6];
  const float* bv = (const float*)d_in[7];
  const float* wp = (const float*)d_in[8];
  const float* bp = (const float*)d_in[9];
  float* out = (float*)d_out;

  char* ws = (char*)d_ws;
  _Float16* xh  = (_Float16*)(ws);                 // 8 MB   f16 x
  _Float16* Qh  = (_Float16*)(ws + 8388608);       // 8 MB   Q (pre-scaled)
  _Float16* Kh  = (_Float16*)(ws + 16777216);      // 8 MB   K
  _Float16* Vt  = (_Float16*)(ws + 25165824);      // 8 MB   V transposed [b][c][n]
  _Float16* Oh  = (_Float16*)(ws + 33554432);      // 8 MB   attention output
  _Float16* WT  = (_Float16*)(ws + 41943040);      // 512 KB transposed f16 weights
  _Float16* WqT = WT;
  _Float16* WkT = WT + 65536;
  _Float16* WvT = WT + 131072;
  _Float16* WpT = WT + 196608;

  prep_kernel<<<(4194304 + 262144 + 255) / 256, 256, 0, stream>>>(
      x, wq, wk, wv, wp, xh, WT);
  qkv_kernel<<<1024, 128, 0, stream>>>(xh, WqT, WkT, WvT, bq, bk, bv, Qh, Kh, Vt);
  flash_kernel<<<dim3(64, 4), 128, 0, stream>>>(Qh, Kh, Vt, Oh);
  proj_kernel<<<1024, 128, 0, stream>>>(Oh, WpT, bp, x, out);
}
